// MultiHeadAttention_33930241639017
// MI455X (gfx1250) — matrix operand order repaired and checked by reading, not compiled, not run on hardware
//
#include <hip/hip_runtime.h>
#include <hip/hip_bf16.h>
#include <math.h>

// ---------------------------------------------------------------------------
// MHA forward for MI455X (gfx1250), bf16 WMMA path, software-pipelined.
// B=4, S=2048, D=1024, H=16, DH=64.
// ---------------------------------------------------------------------------

typedef __bf16 bf16_t;
typedef __attribute__((ext_vector_type(16))) __bf16 v16bf;
typedef __attribute__((ext_vector_type(8)))  __bf16 v8bf;
typedef __attribute__((ext_vector_type(8)))  float  v8f;

#define BB   4
#define SS   2048
#define DD   1024
#define HH   16
#define DHH  64
#define NEGV (-1e20f)

// ---------------------------------------------------------------------------
// helpers
// ---------------------------------------------------------------------------

__device__ __forceinline__ v8f wmma_bf16(v16bf a, v16bf b, v8f c) {
  return __builtin_amdgcn_wmma_f32_16x16x32_bf16(false, a, false, b, (short)0, c,
                                                 false, false);
}

__device__ __forceinline__ v8f zero8() {
  v8f z;
#pragma unroll
  for (int i = 0; i < 8; ++i) z[i] = 0.0f;
  return z;
}

// A-matrix 16x32 bf16 from a row-major f32 tile (ISA 7.12.2 lane layout):
// lane<16: row=lane, K=0..7 & 16..23 ; lane>=16: row=lane-16, K=8..15 & 24..31
__device__ __forceinline__ v16bf load_A_f32(const float* __restrict__ base, int ld) {
  const int lane = threadIdx.x & 31;
  const int row  = lane & 15;
  const int kb   = (lane < 16) ? 0 : 8;
  const float* p = base + (size_t)row * ld + kb;
  float4 f0 = *(const float4*)(p + 0);
  float4 f1 = *(const float4*)(p + 4);
  float4 f2 = *(const float4*)(p + 16);
  float4 f3 = *(const float4*)(p + 20);
  v16bf a;
  a[0]  = (__bf16)f0.x;  a[1]  = (__bf16)f0.y;  a[2]  = (__bf16)f0.z;  a[3]  = (__bf16)f0.w;
  a[4]  = (__bf16)f1.x;  a[5]  = (__bf16)f1.y;  a[6]  = (__bf16)f1.z;  a[7]  = (__bf16)f1.w;
  a[8]  = (__bf16)f2.x;  a[9]  = (__bf16)f2.y;  a[10] = (__bf16)f2.z;  a[11] = (__bf16)f2.w;
  a[12] = (__bf16)f3.x;  a[13] = (__bf16)f3.y;  a[14] = (__bf16)f3.z;  a[15] = (__bf16)f3.w;
  return a;
}

// A-matrix 16x32 bf16 from a row-major bf16 tile (same lane layout).
__device__ __forceinline__ v16bf load_A_bf16(const bf16_t* __restrict__ base, int ld) {
  const int lane = threadIdx.x & 31;
  const int row  = lane & 15;
  const int kb   = (lane < 16) ? 0 : 8;
  const bf16_t* p = base + (size_t)row * ld + kb;
  v8bf lo = *(const v8bf*)(p);
  v8bf hi = *(const v8bf*)(p + 16);
  return __builtin_shufflevector(lo, hi, 0, 1, 2, 3, 4, 5, 6, 7,
                                 8, 9, 10, 11, 12, 13, 14, 15);
}

// B-matrix 32x16 bf16: element (kk, n) at base[n*ld + kk] (K contiguous per
// column). lane<16: col=lane, K=0..7 & 16..23 ; lane>=16: col=lane-16, K=8..15 & 24..31.
__device__ __forceinline__ v16bf load_B16(const bf16_t* __restrict__ base, int ld) {
  const int lane = threadIdx.x & 31;
  const int col  = lane & 15;
  const int kb   = (lane < 16) ? 0 : 8;
  const bf16_t* p = base + (size_t)col * ld + kb;
  v8bf lo = *(const v8bf*)(p);
  v8bf hi = *(const v8bf*)(p + 16);
  return __builtin_shufflevector(lo, hi, 0, 1, 2, 3, 4, 5, 6, 7,
                                 8, 9, 10, 11, 12, 13, 14, 15);
}

// ---------------------------------------------------------------------------
// Kernel 1: weight transpose + f32 -> bf16 convert.  W (k-in, n-out) row-major
// -> Wt[n*D + k] so GEMM B-operands are per-lane contiguous.
// ---------------------------------------------------------------------------
__global__ __launch_bounds__(256) void wt_transpose(const float* __restrict__ W,
                                                    bf16_t* __restrict__ Wt) {
  __shared__ float tile[32][33];
  const int tx = threadIdx.x & 31;
  const int ty = threadIdx.x >> 5;
  const int n0 = blockIdx.x * 32;
  const int k0 = blockIdx.y * 32;
#pragma unroll
  for (int i = ty; i < 32; i += 8)
    tile[i][tx] = W[(size_t)(k0 + i) * DD + n0 + tx];
  __syncthreads();
#pragma unroll
  for (int i = ty; i < 32; i += 8)
    Wt[(size_t)(n0 + i) * DD + k0 + tx] = (__bf16)tile[tx][i];
}

// ---------------------------------------------------------------------------
// Kernel 2: GEMM  out(M=B*S, N=D) = X @ W  via Wt (bf16, N-major).
// One wave owns a 32x64 tile; K-loop unrolled 2x with ping-pong tile buffers
// so loads for step i+1 overlap the 8 WMMAs of step i.
// MODE 0: f32 A, bf16 row-major out. MODE 1: f32 A, V-transposed (B,H,DH,S)
// bf16 out. MODE 2: bf16 A, f32 row-major out.
// ---------------------------------------------------------------------------
struct GTiles { v16bf a0, a1, b[4]; };

template <int MODE>
__global__ __launch_bounds__(128) void gemm_k(const void* __restrict__ Xv,
                                              const bf16_t* __restrict__ Wt,
                                              void* __restrict__ outv) {
  const int w    = threadIdx.x >> 5;
  const int mt   = blockIdx.x >> 2;                  // 0..255
  const int nt   = ((blockIdx.x & 3) << 2) | w;      // 0..15
  const int m0   = mt * 32;
  const int n0   = nt * 64;
  const int lane = threadIdx.x & 31;
  const int col  = lane & 15;
  const int hi   = lane >> 4;

  v8f acc[2][4];
#pragma unroll
  for (int i = 0; i < 2; ++i)
#pragma unroll
    for (int j = 0; j < 4; ++j) acc[i][j] = zero8();

  auto loadT = [&](GTiles& t, int k) {
    if (MODE == 2) {
      const bf16_t* X = (const bf16_t*)Xv;
      t.a0 = load_A_bf16(X + (size_t)m0 * DD + k, DD);
      t.a1 = load_A_bf16(X + (size_t)(m0 + 16) * DD + k, DD);
    } else {
      const float* X = (const float*)Xv;
      t.a0 = load_A_f32(X + (size_t)m0 * DD + k, DD);
      t.a1 = load_A_f32(X + (size_t)(m0 + 16) * DD + k, DD);
    }
#pragma unroll
    for (int j = 0; j < 4; ++j)
      t.b[j] = load_B16(Wt + (size_t)(n0 + j * 16) * DD + k, DD);
  };
  auto stepT = [&](const GTiles& t) {
#pragma unroll
    for (int j = 0; j < 4; ++j) {
      acc[0][j] = wmma_bf16(t.a0, t.b[j], acc[0][j]);
      acc[1][j] = wmma_bf16(t.a1, t.b[j], acc[1][j]);
    }
  };

  GTiles t0, t1;
  loadT(t0, 0);
  for (int k = 0; k < DD; k += 64) {
    loadT(t1, k + 32);
    stepT(t0);
    if (k + 64 < DD) loadT(t0, k + 64);
    stepT(t1);
  }

#pragma unroll
  for (int blk = 0; blk < 2; ++blk)
#pragma unroll
    for (int j = 0; j < 4; ++j)
#pragma unroll
      for (int r = 0; r < 8; ++r) {
        const int m = m0 + blk * 16 + r + hi * 8;    // D layout: M = r + 8*(lane>=16)
        const int n = n0 + j * 16 + col;             //           N = lane & 15
        const float v = acc[blk][j][r];
        if (MODE == 0) {
          ((bf16_t*)outv)[(size_t)m * DD + n] = (__bf16)v;
        } else if (MODE == 1) {
          const int b  = m / SS, s  = m % SS;
          const int hh = n / DHH, dh = n % DHH;
          ((bf16_t*)outv)[((size_t)(b * HH + hh) * DHH + dh) * SS + s] = (__bf16)v;
        } else {
          ((float*)outv)[(size_t)m * DD + n] = v;
        }
      }
}

// ---------------------------------------------------------------------------
// Kernel 3: flash attention.  One wave per (b, h, 16-query tile).
// K tiles double-buffered across 32-key blocks; V tiles issued before the
// score WMMAs so they drain during scores+softmax. Row-sums of P computed
// with an extra WMMA against a ones B-matrix (replaces 32 ds_bpermutes).
// ---------------------------------------------------------------------------
__global__ __launch_bounds__(128) void attn_fa(const bf16_t* __restrict__ Qp,
                                               const bf16_t* __restrict__ Kp,
                                               const bf16_t* __restrict__ Vt,
                                               const long long* __restrict__ vlens,
                                               bf16_t* __restrict__ At) {
  __shared__ alignas(16) bf16_t pbuf[4][16][40];     // per-wave P tile (padded)

  const int w    = threadIdx.x >> 5;
  const int wid  = blockIdx.x * 4 + w;
  const int QT   = SS / 16;
  const int qt   = wid % QT;
  const int bh   = wid / QT;
  const int h    = bh % HH;
  const int b    = bh / HH;
  const int lane = threadIdx.x & 31;
  const int col  = lane & 15;
  const int hi   = lane >> 4;
  const int vlen = (int)vlens[b];

  const bf16_t* Qb = Qp + ((size_t)(b * SS + qt * 16)) * DD + h * DHH;
  const v16bf aq0 = load_A_bf16(Qb, DD);        // dh 0..31
  const v16bf aq1 = load_A_bf16(Qb + 32, DD);   // dh 32..63

  v16bf onesB;                                  // 32x16 all-ones B operand
#pragma unroll
  for (int i = 0; i < 16; ++i) onesB[i] = (__bf16)1.0f;

  v8f o[4];
#pragma unroll
  for (int t = 0; t < 4; ++t) o[t] = zero8();
  float mrun[8], lrun[8];
#pragma unroll
  for (int r = 0; r < 8; ++r) { mrun[r] = -3.0e38f; lrun[r] = 0.0f; }

  const bf16_t* Kb_all = Kp + (size_t)b * SS * DD + h * DHH;
  const bf16_t* Vb_all = Vt + (size_t)(b * HH + h) * DHH * SS;
  const float   scale  = 0.125f;                // 1/sqrt(64)

  auto loadK = [&](v16bf* kt, int kblk) {
    const bf16_t* Kb = Kb_all + (size_t)kblk * DD;
    kt[0] = load_B16(Kb, DD);                       // keys 0..15, dh 0..31
    kt[1] = load_B16(Kb + 32, DD);                  // keys 0..15, dh 32..63
    kt[2] = load_B16(Kb + (size_t)16 * DD, DD);     // keys 16..31, dh 0..31
    kt[3] = load_B16(Kb + (size_t)16 * DD + 32, DD);
  };

  auto process = [&](const v16bf* kt, int kblk) {
    // V tiles for this block: issue loads first, consume after softmax.
    v16bf vt[4];
    const bf16_t* Vb = Vb_all + kblk;
#pragma unroll
    for (int t = 0; t < 4; ++t)
      vt[t] = load_B16(Vb + (size_t)(t * 16) * SS, SS);

    v8f s0 = zero8(), s1 = zero8();
    s0 = wmma_bf16(aq0, kt[0], s0);
    s0 = wmma_bf16(aq1, kt[1], s0);
    s1 = wmma_bf16(aq0, kt[2], s1);
    s1 = wmma_bf16(aq1, kt[3], s1);

    const bool ok0 = (kblk + col) < vlen;
    const bool ok1 = (kblk + 16 + col) < vlen;

    float p0v[8], p1v[8], alpha[8];
#pragma unroll
    for (int r = 0; r < 8; ++r) {
      float x0 = ok0 ? s0[r] * scale : NEGV;
      float x1 = ok1 ? s1[r] * scale : NEGV;
      float rm = fmaxf(x0, x1);                 // row-max over 16 key lanes
      rm = fmaxf(rm, __shfl_xor(rm, 1, 32));
      rm = fmaxf(rm, __shfl_xor(rm, 2, 32));
      rm = fmaxf(rm, __shfl_xor(rm, 4, 32));
      rm = fmaxf(rm, __shfl_xor(rm, 8, 32));
      const float mn = fmaxf(mrun[r], rm);
      alpha[r] = __expf(mrun[r] - mn);
      mrun[r]  = mn;
      p0v[r]   = __expf(x0 - mn);
      p1v[r]   = __expf(x1 - mn);
    }
#pragma unroll
    for (int t = 0; t < 4; ++t)
#pragma unroll
      for (int r = 0; r < 8; ++r) o[t][r] *= alpha[r];

    // D-layout -> A-layout via per-wave LDS tile (same-wave DS ordering).
#pragma unroll
    for (int r = 0; r < 8; ++r) {
      const int row = r + hi * 8;
      pbuf[w][row][col]      = (__bf16)p0v[r];
      pbuf[w][row][16 + col] = (__bf16)p1v[r];
    }
    const int arow = lane & 15;
    const int akb  = (lane < 16) ? 0 : 8;
    const bf16_t* pr = &pbuf[w][arow][0];
    v8bf lo = *(const v8bf*)(pr + akb);
    v8bf hb = *(const v8bf*)(pr + 16 + akb);
    v16bf ap = __builtin_shufflevector(lo, hb, 0, 1, 2, 3, 4, 5, 6, 7,
                                       8, 9, 10, 11, 12, 13, 14, 15);

    // Row sums of P on the matrix pipe: P @ ones -> every lane gets its 8
    // row-sums in D layout (same mapping as lrun).
    v8f rsum = zero8();
    rsum = wmma_bf16(ap, onesB, rsum);
#pragma unroll
    for (int t = 0; t < 4; ++t)
      o[t] = wmma_bf16(ap, vt[t], o[t]);
#pragma unroll
    for (int r = 0; r < 8; ++r) lrun[r] = lrun[r] * alpha[r] + rsum[r];
  };

  // Ping-pong K tiles: load block i+1 while computing block i.
  v16bf ka[4], kb[4];
  loadK(ka, 0);
  for (int kblk = 0; kblk < SS; kblk += 64) {
    loadK(kb, kblk + 32);
    process(ka, kblk);
    if (kblk + 64 < SS) loadK(ka, kblk + 64);
    process(kb, kblk + 32);
  }

#pragma unroll
  for (int t = 0; t < 4; ++t)
#pragma unroll
    for (int r = 0; r < 8; ++r) {
      const int row = r + hi * 8;
      const int s   = qt * 16 + row;
      const int dh  = t * 16 + col;
      const float v = o[t][r] / lrun[r];
      At[((size_t)(b * SS + s)) * DD + h * DHH + dh] = (__bf16)v;
    }
}

// ---------------------------------------------------------------------------
// launch
// ---------------------------------------------------------------------------
extern "C" void kernel_launch(void* const* d_in, const int* in_sizes, int n_in,
                              void* d_out, int out_size, void* d_ws, size_t ws_size,
                              hipStream_t stream) {
  (void)in_sizes; (void)n_in; (void)out_size; (void)ws_size;

  const float*     Q  = (const float*)d_in[0];
  const float*     K  = (const float*)d_in[1];
  const float*     V  = (const float*)d_in[2];
  const long long* vl = (const long long*)d_in[3];
  const float*     Wq = (const float*)d_in[4];
  const float*     Wk = (const float*)d_in[5];
  const float*     Wv = (const float*)d_in[6];
  const float*     Wo = (const float*)d_in[7];
  float*           out = (float*)d_out;

  // workspace: 4x (D*D) bf16 transposed weights + 4x (B*S*D) bf16 = 72 MB
  bf16_t* Wtq = (bf16_t*)d_ws;
  bf16_t* Wtk = Wtq + (size_t)DD * DD;
  bf16_t* Wtv = Wtk + (size_t)DD * DD;
  bf16_t* Wto = Wtv + (size_t)DD * DD;
  bf16_t* Qp  = Wto + (size_t)DD * DD;
  bf16_t* Kp  = Qp + (size_t)BB * SS * DD;
  bf16_t* Vt  = Kp + (size_t)BB * SS * DD;
  bf16_t* At  = Vt + (size_t)BB * SS * DD;

  dim3 tg(DD / 32, DD / 32);
  wt_transpose<<<tg, 256, 0, stream>>>(Wq, Wtq);
  wt_transpose<<<tg, 256, 0, stream>>>(Wk, Wtk);
  wt_transpose<<<tg, 256, 0, stream>>>(Wv, Wtv);
  wt_transpose<<<tg, 256, 0, stream>>>(Wo, Wto);

  const int gemm_blocks = (BB * SS / 32) * 4;        // 1024 blocks x 4 waves
  gemm_k<0><<<gemm_blocks, 128, 0, stream>>>((const void*)Q, Wtq, (void*)Qp);
  gemm_k<0><<<gemm_blocks, 128, 0, stream>>>((const void*)K, Wtk, (void*)Kp);
  gemm_k<1><<<gemm_blocks, 128, 0, stream>>>((const void*)V, Wtv, (void*)Vt);

  const int attn_blocks = BB * HH * (SS / 16) / 4;   // 2048 blocks x 4 waves
  attn_fa<<<attn_blocks, 128, 0, stream>>>(Qp, Kp, Vt, vl, At);

  gemm_k<2><<<gemm_blocks, 128, 0, stream>>>((const void*)At, Wto, (void*)out);
}
